// SF_Gatv2_MLP_69054484185859
// MI455X (gfx1250) — matrix operand order repaired and checked by reading, not compiled, not run on hardware
//
#include <hip/hip_runtime.h>
#include <hip/hip_bf16.h>
#include <cstdint>
#include <cstddef>

// ---------------- problem constants (match reference) ----------------
#define NN      20000
#define EE      640000
#define GG      64
#define F_IN    128
#define F_EDGE  16
#define HID     64
#define HEADS   4
#define OUTF    128
#define IAW_D   200
#define PREDH   256
#define MERGE_D (2*OUTF + IAW_D)   // 456
#define EN      (EE + NN)          // edges + self loops

typedef __attribute__((ext_vector_type(16))) __bf16 v16bf;
typedef __attribute__((ext_vector_type(8)))  float  v8f;

// ---------------- small helpers ----------------
__device__ inline void atomicMaxF(float* addr, float val) {
  unsigned int* ua = (unsigned int*)addr;
  unsigned int old = __float_as_uint(*addr);
  while (__uint_as_float(old) < val) {
    unsigned int assumed = old;
    old = atomicCAS(ua, assumed, __float_as_uint(val));
    if (old == assumed) break;
  }
}

__device__ inline unsigned short f32_to_bf16(float f) {
  unsigned int u = __float_as_uint(f);
  return (unsigned short)((u + 0x7FFFu + ((u >> 16) & 1u)) >> 16);
}

__global__ void fill_kernel(float* p, float v, long n) {
  long i = (long)blockIdx.x * blockDim.x + threadIdx.x;
  if (i < n) p[i] = v;
}

// f32 -> bf16 (round to nearest even), packed as u16
__global__ void cvt_bf16_kernel(const float* __restrict__ in,
                                unsigned short* __restrict__ out, long n) {
  long i = (long)blockIdx.x * blockDim.x + threadIdx.x;
  if (i >= n) return;
  out[i] = f32_to_bf16(in[i]);
}

// ---------------- weight pack: f32 [K,N] -> bf16 WMMA B-fragment layout ----
// packed idx = (((kt*tiles_n + nt)*32) + lane)*16 + j
// element j of lane's fragment = B[kt*32 + (lane>>4)*16 + j][nt*16 + (lane&15)]
// (contiguous k order k = 16*(lane>>4) + j, same as the GEMM's A load; cols across lanes)
__global__ void pack_b_kernel(const float* __restrict__ W,
                              unsigned short* __restrict__ Bp, int K, int N) {
  long total = (long)K * N;
  long i = (long)blockIdx.x * blockDim.x + threadIdx.x;
  if (i >= total) return;
  int j    = (int)(i & 15);
  int lane = (int)((i >> 4) & 31);
  long tile = i >> 9;                 // kt*tiles_n + nt
  int tiles_n = N >> 4;
  int kt = (int)(tile / tiles_n), nt = (int)(tile - (long)kt * tiles_n);
  int row = kt * 32 + (lane >> 4) * 16 + j;
  int col = nt * 16 + (lane & 15);
  Bp[i] = f32_to_bf16(W[(size_t)row * N + col]);
}

// ---------------- WMMA GEMM: C[M,N] = bf16(A[M,K]) @ packed_bf16(B) + bias --
// One wave computes a 16x64 strip (4 accumulators): A fragment loaded once per
// k-step, 4 packed B fragments -> 4 v_wmma per loop body.
// K % 32 == 0, M % 16 == 0, N % 64 == 0.
__global__ void gemm_bf16_wmma_kernel(const unsigned short* __restrict__ A,
                                      const unsigned short* __restrict__ Bp,
                                      const float* __restrict__ bias,
                                      float* __restrict__ C,
                                      int M, int K, int N) {
  int wave = (int)((blockIdx.x * blockDim.x + threadIdx.x) >> 5);
  int lane = threadIdx.x & 31;
  int ngroups = N >> 6;                       // 64-column groups
  if (wave >= (M >> 4) * ngroups) return;     // wave-uniform: EXEC stays all-ones
  int tm = wave / ngroups, tg = wave - tm * ngroups;
  int m0  = tm << 4;
  int tn0 = tg << 2;                          // first 16-col tile of this strip
  int tiles_n = N >> 4;
  int half = lane >> 4;                       // 0: lanes 0-15, 1: lanes 16-31
  int r    = lane & 15;

  v8f acc0 = {}, acc1 = {}, acc2 = {}, acc3 = {};
  const unsigned int* arow32 =
      (const unsigned int*)(A + (size_t)(m0 + r) * K);
  const v16bf* bpv = (const v16bf*)Bp;

  for (int k0 = 0, kt = 0; k0 < K; k0 += 32, ++kt) {
    v16bf a{};
    // A fragment 16x32: lane half 0 holds K {0..15}, half 1 {16..31}
#pragma unroll
    for (int j = 0; j < 8; ++j) {
      int kb = half * 16 + 2 * j;
      unsigned int u = arow32[(k0 + kb) >> 1];
      a[2 * j]     = __builtin_bit_cast(__bf16, (unsigned short)(u & 0xFFFFu));
      a[2 * j + 1] = __builtin_bit_cast(__bf16, (unsigned short)(u >> 16));
    }
    const v16bf* bb = bpv + ((size_t)kt * tiles_n + tn0) * 32 + lane;
    v16bf b0 = bb[0];
    v16bf b1 = bb[32];
    v16bf b2 = bb[64];
    v16bf b3 = bb[96];
    acc0 = __builtin_amdgcn_wmma_f32_16x16x32_bf16(false, a, false, b0, (short)0, acc0, false, false);
    acc1 = __builtin_amdgcn_wmma_f32_16x16x32_bf16(false, a, false, b1, (short)0, acc1, false, false);
    acc2 = __builtin_amdgcn_wmma_f32_16x16x32_bf16(false, a, false, b2, (short)0, acc2, false, false);
    acc3 = __builtin_amdgcn_wmma_f32_16x16x32_bf16(false, a, false, b3, (short)0, acc3, false, false);
  }
  // C/D layout: VGPR v, lanes 0-15 -> M=v ; lanes 16-31 -> M=v+8 ; N = lane&15
  int rowbase = m0 + half * 8;
  v8f accs[4] = {acc0, acc1, acc2, acc3};
#pragma unroll
  for (int t = 0; t < 4; ++t) {
    int col = (tn0 + t) * 16 + r;
    float bv = bias ? bias[col] : 0.0f;
#pragma unroll
    for (int v = 0; v < 8; ++v)
      C[(size_t)(rowbase + v) * N + col] = accs[t][v] + bv;
  }
}

// ---------------- self-loop edge_attr (fill_value='mean') ----------------
__global__ void deg_loop_kernel(const long long* __restrict__ dst,
                                const float* __restrict__ edge_attr,
                                float* deg, float* loop_attr) {
  int i = blockIdx.x * blockDim.x + threadIdx.x;
  if (i >= EE) return;
  int d = (int)dst[i];
  atomicAdd(&deg[d], 1.0f);
  const float* ea = edge_attr + (size_t)i * F_EDGE;
#pragma unroll
  for (int f = 0; f < F_EDGE; ++f) atomicAdd(&loop_attr[(size_t)d * F_EDGE + f], ea[f]);
}

__global__ void loop_norm_kernel(float* loop_attr, const float* deg) {
  int i = blockIdx.x * blockDim.x + threadIdx.x;
  if (i >= NN * F_EDGE) return;
  float dg = deg[i >> 4];
  loop_attr[i] /= (dg > 1.0f ? dg : 1.0f);
}

// ---------------- GATv2 edge phase 1: attention logits + segment max -------
// one wave per edge; lane covers HC/32 contiguous channels (8 lanes per head).
template <int HC>
__global__ void edge_alpha_kernel(const float* __restrict__ xl, const float* __restrict__ xr,
                                  const float* __restrict__ edge_attr,
                                  const float* __restrict__ loop_attr,
                                  const long long* __restrict__ src,
                                  const long long* __restrict__ dst,
                                  const float* __restrict__ We,   // [16, HC]
                                  const float* __restrict__ att,  // [HC]
                                  float* __restrict__ alpha,      // [EN, H]
                                  float* __restrict__ amax) {     // [N, H]
  constexpr int CHUNK = HC / 32;
  int wave = (int)((blockIdx.x * blockDim.x + threadIdx.x) >> 5);
  int lane = threadIdx.x & 31;
  if (wave >= EN) return;
  int e = wave, s, d;
  const float* ea;
  if (e < EE) { s = (int)src[e]; d = (int)dst[e]; ea = edge_attr + (size_t)e * F_EDGE; }
  else        { s = d = e - EE;                   ea = loop_attr + (size_t)s * F_EDGE; }
  float eav[F_EDGE];
#pragma unroll
  for (int f = 0; f < F_EDGE; ++f) eav[f] = ea[f];
  int c0 = lane * CHUNK;
  const float* xls = xl + (size_t)s * HC + c0;
  const float* xrd = xr + (size_t)d * HC + c0;
  float partial = 0.0f;
#pragma unroll
  for (int cc = 0; cc < CHUNK; ++cc) {
    int c = c0 + cc;
    float em = 0.0f;
#pragma unroll
    for (int f = 0; f < F_EDGE; ++f) em = fmaf(eav[f], We[(size_t)f * HC + c], em);
    float m = xls[cc] + xrd[cc] + em;
    float lr = m > 0.0f ? m : 0.2f * m;
    partial = fmaf(lr, att[c], partial);
  }
  // reduce over the 8 lanes of each head (H=4 fixed)
  partial += __shfl_xor(partial, 1, 32);
  partial += __shfl_xor(partial, 2, 32);
  partial += __shfl_xor(partial, 4, 32);
  if ((lane & 7) == 0) {
    int h = lane >> 3;
    alpha[(size_t)e * HEADS + h] = partial;
    atomicMaxF(&amax[(size_t)d * HEADS + h], partial);
  }
}

// phase 2: exp(alpha - max) and segment denominator
__global__ void edge_softmax_kernel(const long long* __restrict__ dst,
                                    float* __restrict__ alpha,      // in logits, out exp
                                    const float* __restrict__ amax,
                                    float* __restrict__ den) {
  long i = (long)blockIdx.x * blockDim.x + threadIdx.x;
  if (i >= (long)EN * HEADS) return;
  int e = (int)(i / HEADS), h = (int)(i - (long)e * HEADS);
  int d = (e < EE) ? (int)dst[e] : (e - EE);
  float v = __expf(alpha[i] - amax[(size_t)d * HEADS + h]);
  alpha[i] = v;
  atomicAdd(&den[(size_t)d * HEADS + h], v);
}

// phase 3: out[dst] += xl[src] * a   (weighted scatter-add)
template <int HC>
__global__ void edge_scatter_kernel(const float* __restrict__ xl,
                                    const float* __restrict__ alpha,
                                    const float* __restrict__ den,
                                    const long long* __restrict__ src,
                                    const long long* __restrict__ dst,
                                    float* __restrict__ outacc) {
  constexpr int CHUNK = HC / 32;
  int wave = (int)((blockIdx.x * blockDim.x + threadIdx.x) >> 5);
  int lane = threadIdx.x & 31;
  if (wave >= EN) return;
  int e = wave, s, d;
  if (e < EE) { s = (int)src[e]; d = (int)dst[e]; } else { s = d = e - EE; }
  int c0 = lane * CHUNK;
  int h = lane >> 3;
  float a = alpha[(size_t)e * HEADS + h] / den[(size_t)d * HEADS + h];
  const float* xls = xl + (size_t)s * HC + c0;
  float* od = outacc + (size_t)d * HC + c0;
#pragma unroll
  for (int cc = 0; cc < CHUNK; ++cc)
    atomicAdd(&od[cc], xls[cc] * a);
}

// phase 4: add bias, optionally mean over heads
__global__ void node_finalize_kernel(const float* __restrict__ outacc,
                                     const float* __restrict__ bias,
                                     float* __restrict__ hout,
                                     int C, int HC, int concat) {
  long total = concat ? (long)NN * HC : (long)NN * C;
  long i = (long)blockIdx.x * blockDim.x + threadIdx.x;
  if (i >= total) return;
  if (concat) {
    int c = (int)(i % HC);
    hout[i] = outacc[i] + bias[c];
  } else {
    int n = (int)(i / C), c = (int)(i - (long)n * C);
    float v = 0.0f;
#pragma unroll
    for (int h = 0; h < HEADS; ++h) v += outacc[(size_t)n * HC + h * C + c];
    hout[i] = v * (1.0f / HEADS) + bias[c];
  }
}

// ---------------- global pooling (max + mean per graph) ----------------
__global__ void pool_kernel(const float* __restrict__ h, const long long* __restrict__ batch,
                            float* pmax, float* psum, float* cnt) {
  long i = (long)blockIdx.x * blockDim.x + threadIdx.x;
  if (i >= (long)NN * OUTF) return;
  int n = (int)(i / OUTF), f = (int)(i - (long)n * OUTF);
  int g = (int)batch[n];
  float v = h[i];
  atomicMaxF(&pmax[(size_t)g * OUTF + f], v);
  atomicAdd(&psum[(size_t)g * OUTF + f], v);
  if (f == 0) atomicAdd(&cnt[g], 1.0f);
}

// ---------------- small dense layers (64-row MLPs) ----------------
// act: 0 = none, 1 = leaky_relu(0.2), 2 = elu
__global__ void dense_kernel(const float* __restrict__ in, const float* __restrict__ W,
                             const float* __restrict__ b, float* __restrict__ out,
                             int M, int K, int Nc, int act) {
  int i = blockIdx.x * blockDim.x + threadIdx.x;
  if (i >= M * Nc) return;
  int m = i / Nc, j = i - m * Nc;
  float acc = b ? b[j] : 0.0f;
  const float* row = in + (size_t)m * K;
  for (int k = 0; k < K; ++k) acc = fmaf(row[k], W[(size_t)k * Nc + j], acc);
  if (act == 1) acc = acc > 0.0f ? acc : 0.2f * acc;
  else if (act == 2) acc = acc > 0.0f ? acc : (__expf(acc) - 1.0f);
  out[i] = acc;
}

__global__ void layernorm_kernel(const float* __restrict__ t, const float* __restrict__ gam,
                                 const float* __restrict__ bet, float* __restrict__ out, int D) {
  __shared__ float red[256];
  int g = blockIdx.x, tid = threadIdx.x;
  float s = 0.0f;
  for (int j = tid; j < D; j += 256) s += t[(size_t)g * D + j];
  red[tid] = s; __syncthreads();
  for (int off = 128; off; off >>= 1) { if (tid < off) red[tid] += red[tid + off]; __syncthreads(); }
  float mu = red[0] / D; __syncthreads();
  float s2 = 0.0f;
  for (int j = tid; j < D; j += 256) { float dlt = t[(size_t)g * D + j] - mu; s2 = fmaf(dlt, dlt, s2); }
  red[tid] = s2; __syncthreads();
  for (int off = 128; off; off >>= 1) { if (tid < off) red[tid] += red[tid + off]; __syncthreads(); }
  float inv = rsqrtf(red[0] / D + 1e-5f);
  for (int j = tid; j < D; j += 256)
    out[(size_t)g * D + j] = (t[(size_t)g * D + j] - mu) * inv * gam[j] + bet[j];
}

__global__ void mf_kernel(const float* pmax, const float* psum, const float* cnt,
                          const float* iawf, float* mf) {
  int i = blockIdx.x * blockDim.x + threadIdx.x;
  if (i >= GG * MERGE_D) return;
  int g = i / MERGE_D, f = i - g * MERGE_D;
  float v;
  if (f < OUTF) v = pmax[(size_t)g * OUTF + f];
  else if (f < 2 * OUTF) {
    float c = cnt[g]; c = c > 1.0f ? c : 1.0f;
    v = psum[(size_t)g * OUTF + (f - OUTF)] / c;
  } else v = iawf[(size_t)g * IAW_D + (f - 2 * OUTF)];
  mf[i] = v;
}

// ---------------- host orchestration ----------------
static inline int grid1(long n, int blk) { return (int)((n + blk - 1) / blk); }

extern "C" void kernel_launch(void* const* d_in, const int* in_sizes, int n_in,
                              void* d_out, int out_size, void* d_ws, size_t ws_size,
                              hipStream_t stream) {
  (void)in_sizes; (void)n_in; (void)out_size; (void)ws_size;
  const int BLK = 256;

  // -------- input pointers (setup_inputs dict order; params = pytree order:
  // dict keys sorted -> gat, iaw, merge, pred; gat dict keys sorted:
  // We, Wl, Wr, att, bias, bl, br; tuples in order) ----
  const float*     x         = (const float*)d_in[0];
  const long long* edge_idx  = (const long long*)d_in[1];
  const long long* srcI      = edge_idx;
  const long long* dstI      = edge_idx + EE;
  const float*     edge_attr = (const float*)d_in[2];
  const float*     iaw_attr  = (const float*)d_in[3];
  const long long* batch     = (const long long*)d_in[4];

  struct GatP { const float *We, *Wl, *Wr, *att, *bias, *bl, *br; };
  GatP gp[3];
  int base = 5;
  for (int l = 0; l < 3; ++l) {
    gp[l].We   = (const float*)d_in[base + 0];
    gp[l].Wl   = (const float*)d_in[base + 1];
    gp[l].Wr   = (const float*)d_in[base + 2];
    gp[l].att  = (const float*)d_in[base + 3];
    gp[l].bias = (const float*)d_in[base + 4];
    gp[l].bl   = (const float*)d_in[base + 5];
    gp[l].br   = (const float*)d_in[base + 6];
    base += 7;
  }
  const float *iW1 = (const float*)d_in[26], *ib1 = (const float*)d_in[27];
  const float *iW2 = (const float*)d_in[28], *ib2 = (const float*)d_in[29];
  const float *igam = (const float*)d_in[30], *ibet = (const float*)d_in[31];
  const float *Wm1 = (const float*)d_in[32], *bm1 = (const float*)d_in[33];
  const float *Wm2 = (const float*)d_in[34], *bm2 = (const float*)d_in[35];
  const float *Wp1 = (const float*)d_in[36], *bp1 = (const float*)d_in[37];
  const float *Wp2 = (const float*)d_in[38], *bp2 = (const float*)d_in[39];
  const float *Wp3 = (const float*)d_in[40], *bp3 = (const float*)d_in[41];

  // -------- workspace bump allocator (bytes, 256B aligned chunks) --------
  char* wsp = (char*)d_ws;
  size_t off = 0;
  auto alloc = [&](size_t bytes) {
    void* p = wsp + off;
    off += (bytes + 255) & ~(size_t)255;
    return p;
  };
  float* deg       = (float*)alloc(sizeof(float) * NN);
  float* loop_attr = (float*)alloc(sizeof(float) * NN * F_EDGE);
  unsigned short* h_bf   = (unsigned short*)alloc(sizeof(unsigned short) * (size_t)NN * 256);
  unsigned short* w_pack = (unsigned short*)alloc(sizeof(unsigned short) * 256 * 512);
  float* xl     = (float*)alloc(sizeof(float) * (size_t)NN * 512);
  float* xr     = (float*)alloc(sizeof(float) * (size_t)NN * 512);
  float* alpha  = (float*)alloc(sizeof(float) * (size_t)EN * HEADS);
  float* amax   = (float*)alloc(sizeof(float) * (size_t)NN * HEADS);
  float* den    = (float*)alloc(sizeof(float) * (size_t)NN * HEADS);
  float* outacc = (float*)alloc(sizeof(float) * (size_t)NN * 512);
  float* h_buf  = (float*)alloc(sizeof(float) * (size_t)NN * 256);
  float* pmax   = (float*)alloc(sizeof(float) * GG * OUTF);
  float* psum   = (float*)alloc(sizeof(float) * GG * OUTF);
  float* cnt    = (float*)alloc(sizeof(float) * GG);
  float* t1     = (float*)alloc(sizeof(float) * GG * IAW_D);
  float* t2     = (float*)alloc(sizeof(float) * GG * IAW_D);
  float* iawf   = (float*)alloc(sizeof(float) * GG * IAW_D);
  float* mf     = (float*)alloc(sizeof(float) * GG * MERGE_D);
  float* mf2    = (float*)alloc(sizeof(float) * GG * MERGE_D);
  float* mf3    = (float*)alloc(sizeof(float) * GG * MERGE_D);
  float* p1     = (float*)alloc(sizeof(float) * GG * PREDH);
  float* p2     = (float*)alloc(sizeof(float) * GG * PREDH);

  // -------- self-loop attrs: mean of incoming edge_attr per node --------
  fill_kernel<<<grid1(NN, BLK), BLK, 0, stream>>>(deg, 0.0f, NN);
  fill_kernel<<<grid1((long)NN * F_EDGE, BLK), BLK, 0, stream>>>(loop_attr, 0.0f, (long)NN * F_EDGE);
  deg_loop_kernel<<<grid1(EE, BLK), BLK, 0, stream>>>(dstI, edge_attr, deg, loop_attr);
  loop_norm_kernel<<<grid1((long)NN * F_EDGE, BLK), BLK, 0, stream>>>(loop_attr, deg);

  // -------- 3 GATv2 layers --------
  const int lay_din[3]    = {F_IN, HID * HEADS, HID * HEADS};  // 128, 256, 256
  const int lay_C[3]      = {HID, HID, OUTF};                  // 64, 64, 128
  const int lay_concat[3] = {1, 1, 0};
  const float* hcur = x;

  for (int l = 0; l < 3; ++l) {
    int din = lay_din[l];
    int C   = lay_C[l];
    int HC  = HEADS * C;   // 256, 256, 512

    // node transforms via WMMA GEMM on bf16 (16x64 strip per wave)
    cvt_bf16_kernel<<<grid1((long)NN * din, BLK), BLK, 0, stream>>>(hcur, h_bf, (long)NN * din);

    long waves = (long)(NN / 16) * (HC / 64);
    int gblk = grid1(waves * 32, BLK);  // 8 waves per block
    pack_b_kernel<<<grid1((long)din * HC, BLK), BLK, 0, stream>>>(gp[l].Wl, w_pack, din, HC);
    gemm_bf16_wmma_kernel<<<gblk, BLK, 0, stream>>>(h_bf, w_pack, gp[l].bl, xl, NN, din, HC);
    pack_b_kernel<<<grid1((long)din * HC, BLK), BLK, 0, stream>>>(gp[l].Wr, w_pack, din, HC);
    gemm_bf16_wmma_kernel<<<gblk, BLK, 0, stream>>>(h_bf, w_pack, gp[l].br, xr, NN, din, HC);

    // attention softmax over incoming edges + aggregation
    fill_kernel<<<grid1((long)NN * HEADS, BLK), BLK, 0, stream>>>(amax, -1e30f, (long)NN * HEADS);
    fill_kernel<<<grid1((long)NN * HEADS, BLK), BLK, 0, stream>>>(den, 0.0f, (long)NN * HEADS);
    fill_kernel<<<grid1((long)NN * HC, BLK), BLK, 0, stream>>>(outacc, 0.0f, (long)NN * HC);

    if (HC == 256) {
      edge_alpha_kernel<256><<<grid1((long)EN * 32, BLK), BLK, 0, stream>>>(
          xl, xr, edge_attr, loop_attr, srcI, dstI, gp[l].We, gp[l].att, alpha, amax);
    } else {
      edge_alpha_kernel<512><<<grid1((long)EN * 32, BLK), BLK, 0, stream>>>(
          xl, xr, edge_attr, loop_attr, srcI, dstI, gp[l].We, gp[l].att, alpha, amax);
    }
    edge_softmax_kernel<<<grid1((long)EN * HEADS, BLK), BLK, 0, stream>>>(dstI, alpha, amax, den);
    if (HC == 256) {
      edge_scatter_kernel<256><<<grid1((long)EN * 32, BLK), BLK, 0, stream>>>(
          xl, alpha, den, srcI, dstI, outacc);
    } else {
      edge_scatter_kernel<512><<<grid1((long)EN * 32, BLK), BLK, 0, stream>>>(
          xl, alpha, den, srcI, dstI, outacc);
    }

    long outn = lay_concat[l] ? (long)NN * HC : (long)NN * C;
    node_finalize_kernel<<<grid1(outn, BLK), BLK, 0, stream>>>(
        outacc, gp[l].bias, h_buf, C, HC, lay_concat[l]);
    hcur = h_buf;
  }

  // -------- per-graph max/mean pooling --------
  fill_kernel<<<grid1(GG * OUTF, BLK), BLK, 0, stream>>>(pmax, -1e30f, GG * OUTF);
  fill_kernel<<<grid1(GG * OUTF, BLK), BLK, 0, stream>>>(psum, 0.0f, GG * OUTF);
  fill_kernel<<<grid1(GG, BLK), BLK, 0, stream>>>(cnt, 0.0f, GG);
  pool_kernel<<<grid1((long)NN * OUTF, BLK), BLK, 0, stream>>>(hcur, batch, pmax, psum, cnt);

  // -------- iaw_embed: Linear+LeakyReLU, Linear, LayerNorm --------
  dense_kernel<<<grid1(GG * IAW_D, BLK), BLK, 0, stream>>>(iaw_attr, iW1, ib1, t1, GG, IAW_D, IAW_D, 1);
  dense_kernel<<<grid1(GG * IAW_D, BLK), BLK, 0, stream>>>(t1, iW2, ib2, t2, GG, IAW_D, IAW_D, 0);
  layernorm_kernel<<<GG, 256, 0, stream>>>(t2, igam, ibet, iawf, IAW_D);

  // -------- merge + pred MLP --------
  mf_kernel<<<grid1(GG * MERGE_D, BLK), BLK, 0, stream>>>(pmax, psum, cnt, iawf, mf);
  dense_kernel<<<grid1(GG * MERGE_D, BLK), BLK, 0, stream>>>(mf, Wm1, bm1, mf2, GG, MERGE_D, MERGE_D, 1);
  dense_kernel<<<grid1(GG * MERGE_D, BLK), BLK, 0, stream>>>(mf2, Wm2, bm2, mf3, GG, MERGE_D, MERGE_D, 0);
  dense_kernel<<<grid1(GG * PREDH, BLK), BLK, 0, stream>>>(mf3, Wp1, bp1, p1, GG, MERGE_D, PREDH, 2);
  dense_kernel<<<grid1(GG * PREDH, BLK), BLK, 0, stream>>>(p1, Wp2, bp2, p2, GG, PREDH, PREDH, 2);
  dense_kernel<<<grid1(GG, BLK), BLK, 0, stream>>>(p2, Wp3, bp3, (float*)d_out, GG, PREDH, 1, 0);
}
